// SVTRBlock_43087111914328
// MI455X (gfx1250) — matrix operand order repaired and checked by reading, not compiled, not run on hardware
//
#include <hip/hip_runtime.h>
#include <hip/hip_bf16.h>

typedef __attribute__((ext_vector_type(16))) _Float16 v16h;
typedef __attribute__((ext_vector_type(8)))  _Float16 v8h;
typedef __attribute__((ext_vector_type(8)))  float    v8f;

// ---------------------------------------------------------------------------
// Problem constants (B=8, N=1024, D=768, heads=12, hd=64, H=3072)
// ---------------------------------------------------------------------------
#define BATCH 8
#define SEQ   1024
#define DIM   768
#define NH    12
#define HD    64
#define HDIM  3072
#define MROWS (BATCH * SEQ)   // 8192
#define QKVC  (3 * DIM)       // 2304

union FragU { v16h v; v8h h2[2]; };

__device__ __forceinline__ v8f wmma_f16(v16h a, v16h b, v8f c) {
  return __builtin_amdgcn_wmma_f32_16x16x32_f16(false, a, false, b, (short)0, c, false, false);
}

// Async global -> LDS, 16 bytes per lane (ASYNCcnt-tracked, no VGPR data).
// LDS address for VDST = low 32 bits of the generic shared pointer (LDS aperture
// truncation, ISA 10.2).
#define ASYNC_COPY_B128(gsrc, ldst)                                        \
  asm volatile("global_load_async_to_lds_b128 %0, %1, off"                 \
               :: "v"((unsigned)(size_t)(ldst)), "v"(gsrc) : "memory")
#define ASYNC_WAIT_3() asm volatile("s_wait_asynccnt 3" ::: "memory")
#define ASYNC_WAIT_0() asm volatile("s_wait_asynccnt 0" ::: "memory")

// ---------------------------------------------------------------------------
// Elementwise converters / transposers
// ---------------------------------------------------------------------------
__global__ void __launch_bounds__(256) k_f32_to_f16(const float* __restrict__ in,
                                                    _Float16* __restrict__ out, size_t n) {
  size_t i = (size_t)blockIdx.x * 256 + threadIdx.x;
  if (i < n) out[i] = (_Float16)in[i];
}

// out[c*R + r] = (f16) in[r*C + c]   (weight [R,C] -> Wt [C,R])
__global__ void __launch_bounds__(256) k_transpose_f32_to_f16(const float* __restrict__ in,
                                                              _Float16* __restrict__ out,
                                                              int R, int C) {
  size_t i = (size_t)blockIdx.x * 256 + threadIdx.x;
  if (i < (size_t)R * C) {
    int r = (int)(i / C), c = (int)(i % C);
    out[(size_t)c * R + r] = (_Float16)in[i];
  }
}

// vt[((b*NH + h)*HD + d)*SEQ + n] = qkvh[(b*SEQ + n)*QKVC + 2*DIM + h*HD + d]
__global__ void __launch_bounds__(256) k_build_vt(const _Float16* __restrict__ qkvh,
                                                  _Float16* __restrict__ vt) {
  size_t i = (size_t)blockIdx.x * 256 + threadIdx.x;
  size_t total = (size_t)BATCH * NH * HD * SEQ;
  if (i >= total) return;
  int n = (int)(i & (SEQ - 1));
  int d = (int)((i >> 10) & (HD - 1));
  int h = (int)((i >> 16) % NH);
  int b = (int)(i / ((size_t)SEQ * HD * NH));
  vt[i] = qkvh[((size_t)b * SEQ + n) * QKVC + 2 * DIM + h * HD + d];
}

// ---------------------------------------------------------------------------
// WMMA GEMM with async-LDS double buffering:
//   C[M,N] = A[M,K](f16) @ Wt[N,K]^T(f16) + bias, fused epilogue
// block: 256 thr = 8 waves; wave tile 16x64; block tile 64x128
// Per K-step: A tile 64x32 (4KB) + B tile 128x32 (8KB) staged via
// global_load_async_to_lds_b128, double buffered (24KB LDS).
// ---------------------------------------------------------------------------
#define EPI_QKV   0   // scale cols < DIM by hd^-0.5, store f16
#define EPI_GELU  1   // exact gelu, store f16
#define EPI_RES   2   // + resid (f32), store f32

template <int EPI>
__global__ void __launch_bounds__(256)
k_gemm_wmma(const _Float16* __restrict__ A, const _Float16* __restrict__ Wt,
            const float* __restrict__ bias, int M, int N, int K,
            float* __restrict__ Cf, _Float16* __restrict__ Ch,
            const float* __restrict__ resid) {
  __shared__ __align__(16) _Float16 Abuf[2][64 * 32];
  __shared__ __align__(16) _Float16 Bbuf[2][128 * 32];

  const int tid  = threadIdx.x;
  const int lane = tid & 31;
  const int wave = tid >> 5;
  const int mblk = blockIdx.x * 64;
  const int nblk = blockIdx.y * 128;
  const int mw = (wave & 3) * 16;    // wave row offset within block tile
  const int nw = (wave >> 2) * 64;   // wave col offset within block tile

  // staging maps: A = 64 rows x 4 chunks(16B); B = 128 rows x 2 chunks(32B)
  const int ar = tid >> 2, ac = (tid & 3) << 3;
  const int br = tid >> 1, bc = (tid & 1) << 4;
  const _Float16* gA = A  + (size_t)(mblk + ar) * K + ac;
  const _Float16* gB = Wt + (size_t)(nblk + br) * K + bc;

  const int colBase = lane & 15;
  const int rowOff  = (lane >> 4) * 8;
  if (EPI == EPI_RES)   // warm the residual tile while the K-loop runs
    __builtin_prefetch(resid + (size_t)(mblk + mw + rowOff) * N + nblk + nw + colBase, 0, 1);

  // prologue: stage buffer 0
  ASYNC_COPY_B128(gA,     &Abuf[0][ar * 32 + ac]);
  ASYNC_COPY_B128(gB,     &Bbuf[0][br * 32 + bc]);
  ASYNC_COPY_B128(gB + 8, &Bbuf[0][br * 32 + bc + 8]);

  v8f acc[4] = {};
  for (int k0 = 0; k0 < K; k0 += 32) {
    const int cur = (k0 >> 5) & 1;
    if (k0 + 32 < K) {
      const int nxt = cur ^ 1;
      ASYNC_COPY_B128(gA + k0 + 32, &Abuf[nxt][ar * 32 + ac]);
      ASYNC_COPY_B128(gB + k0 + 32, &Bbuf[nxt][br * 32 + bc]);
      ASYNC_COPY_B128(gB + k0 + 40, &Bbuf[nxt][br * 32 + bc + 8]);
      ASYNC_WAIT_3();              // current buffer's 3 ops complete (in-order)
    } else {
      ASYNC_WAIT_0();
    }
    __syncthreads();

    FragU a;
    const _Float16* ap = &Abuf[cur][(mw + (lane & 15)) * 32 + ((lane >> 4) << 3)];
    a.h2[0] = *(const v8h*)ap;  a.h2[1] = *(const v8h*)(ap + 16);
#pragma unroll
    for (int t = 0; t < 4; ++t) {
      FragU b;
      const _Float16* bp = &Bbuf[cur][(nw + t * 16 + (lane & 15)) * 32 + ((lane >> 4) << 3)];
      b.h2[0] = *(const v8h*)bp;  b.h2[1] = *(const v8h*)(bp + 16);
      acc[t] = wmma_f16(a.v, b.v, acc[t]);
    }
    __syncthreads();               // all waves done reading before next overwrite
  }

  const int m0 = mblk + mw;
  const int n0 = nblk + nw;
#pragma unroll
  for (int t = 0; t < 4; ++t) {
    int n = n0 + t * 16 + colBase;
    float bv = bias[n];
#pragma unroll
    for (int i = 0; i < 8; ++i) {
      int m = m0 + rowOff + i;
      float v = acc[t][i] + bv;
      size_t idx = (size_t)m * N + n;
      if (EPI == EPI_QKV) {
        if (n < DIM) v *= 0.125f;          // hd^-0.5 = 1/8, applied to q
        Ch[idx] = (_Float16)v;
      } else if (EPI == EPI_GELU) {
        float g = 0.5f * v * (1.0f + erff(v * 0.70710678118f));  // exact gelu
        Ch[idx] = (_Float16)g;
      } else {
        Cf[idx] = v + resid[idx];
      }
    }
  }
}

// ---------------------------------------------------------------------------
// Attention: one block per (b, h, 16-row q tile).  8 waves.
// Phase 1: S[16,1024] = (q*scale) @ k^T via WMMA -> LDS (q frags hoisted)
// Phase 2: row softmax (16 lanes/row, shfl_xor), exp -> LDS f16, rowsum kept
// Phase 3: O[16,64] = P @ V via WMMA (waves 0..3, one 16-col d tile each)
// ---------------------------------------------------------------------------
__global__ void __launch_bounds__(256)
k_attention(const _Float16* __restrict__ qkvh, const _Float16* __restrict__ vt,
            _Float16* __restrict__ attn) {
  extern __shared__ char smem[];
  float*    S    = (float*)smem;                                  // 16*1024 f32 = 64KB
  _Float16* P    = (_Float16*)(smem + 16 * SEQ * 4);              // 16*1024 f16 = 32KB
  float*    rsum = (float*)(smem + 16 * SEQ * 4 + 16 * SEQ * 2);  // 16 f32

  const int tid  = threadIdx.x;
  const int lane = tid & 31;
  const int wave = tid >> 5;
  const int mt = blockIdx.x & 63;
  const int h  = (blockIdx.x >> 6) % NH;
  const int b  = blockIdx.x / (64 * NH);
  const int m0 = mt * 16;

  const _Float16* qbase = qkvh + (size_t)b * SEQ * QKVC + (size_t)h * HD;
  const _Float16* kbase = qbase + DIM;

  // -------- phase 1: scores (q fragments loaded once, reused for 8 tiles) ----
  FragU qa[2];
#pragma unroll
  for (int kq = 0; kq < 2; ++kq) {
    const _Float16* ap = qbase + (size_t)(m0 + (lane & 15)) * QKVC + kq * 32 + ((lane >> 4) << 3);
    qa[kq].h2[0] = *(const v8h*)ap;
    qa[kq].h2[1] = *(const v8h*)(ap + 16);
  }
  for (int ct = wave * 8; ct < wave * 8 + 8; ++ct) {
    v8f acc = {};
    const int n0 = ct * 16;
#pragma unroll
    for (int kk = 0; kk < 2; ++kk) {
      FragU ub;
      const _Float16* bp = kbase + (size_t)(n0 + (lane & 15)) * QKVC + kk * 32 + ((lane >> 4) << 3);
      ub.h2[0] = *(const v8h*)bp;  ub.h2[1] = *(const v8h*)(bp + 16);
      acc = wmma_f16(qa[kk].v, ub.v, acc);
    }
    const int c  = n0 + (lane & 15);
    const int r0 = (lane >> 4) * 8;
#pragma unroll
    for (int i = 0; i < 8; ++i) S[(r0 + i) * SEQ + c] = acc[i];
  }
  __syncthreads();

  // -------- phase 2: softmax (row = tid/16, 16 lanes per row) --------
  {
    const int row = tid >> 4;
    const int sl  = tid & 15;
    float mx = -3.4e38f;
    for (int c = sl; c < SEQ; c += 16) mx = fmaxf(mx, S[row * SEQ + c]);
#pragma unroll
    for (int off = 8; off > 0; off >>= 1) mx = fmaxf(mx, __shfl_xor(mx, off, 16));
    float sum = 0.f;
    for (int c = sl; c < SEQ; c += 16) {
      float e = __expf(S[row * SEQ + c] - mx);
      P[row * SEQ + c] = (_Float16)e;
      sum += e;
    }
#pragma unroll
    for (int off = 8; off > 0; off >>= 1) sum += __shfl_xor(sum, off, 16);
    if (sl == 0) rsum[row] = sum;
  }
  __syncthreads();

  // -------- phase 3: O = P @ V --------
  if (wave < 4) {
    const int dt = wave;
    v8f acc = {};
    const _Float16* vbase = vt + ((size_t)(b * NH + h) * HD + dt * 16) * SEQ;
    for (int kk = 0; kk < SEQ; kk += 32) {
      FragU ua, ub;
      const _Float16* ap = P + (lane & 15) * SEQ + kk + ((lane >> 4) << 3);
      ua.h2[0] = *(const v8h*)ap;  ua.h2[1] = *(const v8h*)(ap + 16);
      const _Float16* bp = vbase + (size_t)(lane & 15) * SEQ + kk + ((lane >> 4) << 3);
      ub.h2[0] = *(const v8h*)bp;  ub.h2[1] = *(const v8h*)(bp + 16);
      acc = wmma_f16(ua.v, ub.v, acc);
    }
    const int d  = dt * 16 + (lane & 15);
    const int r0 = (lane >> 4) * 8;
#pragma unroll
    for (int i = 0; i < 8; ++i) {
      int m = r0 + i;
      float o = acc[i] / rsum[m];
      attn[((size_t)(b * SEQ) + m0 + m) * DIM + h * HD + d] = (_Float16)o;
    }
  }
}

// ---------------------------------------------------------------------------
// LayerNorm over D=768: one block(256) per row; optional f16 copy of output
// ---------------------------------------------------------------------------
__global__ void __launch_bounds__(256)
k_layernorm(const float* __restrict__ in, const float* __restrict__ g,
            const float* __restrict__ bta, float* __restrict__ outf,
            _Float16* __restrict__ outh) {
  __shared__ float red[8];
  __shared__ float s_mu, s_rs;
  const int row = blockIdx.x;
  const int tid = threadIdx.x;
  const int lane = tid & 31, wv = tid >> 5;
  const float* x = in + (size_t)row * DIM;
  float v0 = x[tid], v1 = x[tid + 256], v2 = x[tid + 512];

  float s = v0 + v1 + v2;
#pragma unroll
  for (int off = 16; off > 0; off >>= 1) s += __shfl_xor(s, off, 32);
  if (lane == 0) red[wv] = s;
  __syncthreads();
  if (tid == 0) {
    float t = 0.f;
#pragma unroll
    for (int i = 0; i < 8; ++i) t += red[i];
    s_mu = t * (1.0f / DIM);
  }
  __syncthreads();
  const float mu = s_mu;
  float d0 = v0 - mu, d1 = v1 - mu, d2 = v2 - mu;
  float q = d0 * d0 + d1 * d1 + d2 * d2;
#pragma unroll
  for (int off = 16; off > 0; off >>= 1) q += __shfl_xor(q, off, 32);
  __syncthreads();             // protect red[] reuse
  if (lane == 0) red[wv] = q;
  __syncthreads();
  if (tid == 0) {
    float t = 0.f;
#pragma unroll
    for (int i = 0; i < 8; ++i) t += red[i];
    s_rs = rsqrtf(t * (1.0f / DIM) + 1e-6f);
  }
  __syncthreads();
  const float rs = s_rs;
  float o0 = d0 * rs * g[tid]       + bta[tid];
  float o1 = d1 * rs * g[tid + 256] + bta[tid + 256];
  float o2 = d2 * rs * g[tid + 512] + bta[tid + 512];
  float* o = outf + (size_t)row * DIM;
  o[tid] = o0; o[tid + 256] = o1; o[tid + 512] = o2;
  if (outh) {
    _Float16* oh = outh + (size_t)row * DIM;
    oh[tid] = (_Float16)o0; oh[tid + 256] = (_Float16)o1; oh[tid + 512] = (_Float16)o2;
  }
}

// ---------------------------------------------------------------------------
extern "C" void kernel_launch(void* const* d_in, const int* in_sizes, int n_in,
                              void* d_out, int out_size, void* d_ws, size_t ws_size,
                              hipStream_t stream) {
  const float* x      = (const float*)d_in[0];
  const float* qkv_w  = (const float*)d_in[1];
  const float* qkv_b  = (const float*)d_in[2];
  const float* proj_w = (const float*)d_in[3];
  const float* proj_b = (const float*)d_in[4];
  const float* ln1_g  = (const float*)d_in[5];
  const float* ln1_b  = (const float*)d_in[6];
  const float* ln2_g  = (const float*)d_in[7];
  const float* ln2_b  = (const float*)d_in[8];
  const float* fc1_w  = (const float*)d_in[9];
  const float* fc1_b  = (const float*)d_in[10];
  const float* fc2_w  = (const float*)d_in[11];
  const float* fc2_b  = (const float*)d_in[12];
  float* out = (float*)d_out;

  char* ws = (char*)d_ws;
  size_t off = 0;
  auto carve = [&](size_t bytes) -> char* {
    char* p = ws + off;
    off += (bytes + 255) & ~(size_t)255;
    return p;
  };
  _Float16* xh      = (_Float16*)carve((size_t)MROWS * DIM * 2);     // x f16; reused for attn out
  _Float16* qkv_wt  = (_Float16*)carve((size_t)DIM * QKVC * 2);
  _Float16* proj_wt = (_Float16*)carve((size_t)DIM * DIM * 2);
  _Float16* fc1_wt  = (_Float16*)carve((size_t)DIM * HDIM * 2);
  _Float16* fc2_wt  = (_Float16*)carve((size_t)HDIM * DIM * 2);
  _Float16* qkvh    = (_Float16*)carve((size_t)MROWS * QKVC * 2);
  _Float16* vt      = (_Float16*)carve((size_t)BATCH * NH * HD * SEQ * 2);
  float*    t1      = (float*)carve((size_t)MROWS * DIM * 4);        // resid1 out; reused for resid2 out
  float*    xn      = (float*)carve((size_t)MROWS * DIM * 4);        // ln1 out f32
  _Float16* xnh     = (_Float16*)carve((size_t)MROWS * DIM * 2);     // ln1 out f16
  _Float16* h1      = (_Float16*)carve((size_t)MROWS * HDIM * 2);    // gelu(fc1) f16

  // 0) precision staging: x -> f16; weights -> f16 transposed [N,K]
  {
    size_t n = (size_t)MROWS * DIM;
    k_f32_to_f16<<<(unsigned)((n + 255) / 256), 256, 0, stream>>>(x, xh, n);
  }
  k_transpose_f32_to_f16<<<(DIM * QKVC + 255) / 256, 256, 0, stream>>>(qkv_w, qkv_wt, DIM, QKVC);
  k_transpose_f32_to_f16<<<(DIM * DIM + 255) / 256, 256, 0, stream>>>(proj_w, proj_wt, DIM, DIM);
  k_transpose_f32_to_f16<<<(DIM * HDIM + 255) / 256, 256, 0, stream>>>(fc1_w, fc1_wt, DIM, HDIM);
  k_transpose_f32_to_f16<<<(HDIM * DIM + 255) / 256, 256, 0, stream>>>(fc2_w, fc2_wt, HDIM, DIM);

  // 1) QKV = x @ qkv_w + b   (q pre-scaled by hd^-0.5), f16
  k_gemm_wmma<EPI_QKV><<<dim3(MROWS / 64, QKVC / 128), 256, 0, stream>>>(
      xh, qkv_wt, qkv_b, MROWS, QKVC, DIM, nullptr, qkvh, nullptr);

  // 2) V transpose per head -> vt[b,h,d,n]
  {
    size_t n = (size_t)BATCH * NH * HD * SEQ;
    k_build_vt<<<(unsigned)((n + 255) / 256), 256, 0, stream>>>(qkvh, vt);
  }

  // 3) attention -> attn f16 (reuse xh buffer)
  {
    const size_t shmem = (size_t)16 * SEQ * 4 + (size_t)16 * SEQ * 2 + 64 * 4;
    k_attention<<<BATCH * NH * 64, 256, shmem, stream>>>(qkvh, vt, xh);
  }

  // 4) t1 = x + attn @ proj_w + proj_b
  k_gemm_wmma<EPI_RES><<<dim3(MROWS / 64, DIM / 128), 256, 0, stream>>>(
      xh, proj_wt, proj_b, MROWS, DIM, DIM, t1, nullptr, x);

  // 5) xn = LN1(t1)  (f32 + f16 copy)
  k_layernorm<<<MROWS, 256, 0, stream>>>(t1, ln1_g, ln1_b, xn, xnh);

  // 6) h1 = gelu(xn @ fc1_w + fc1_b)  f16
  k_gemm_wmma<EPI_GELU><<<dim3(MROWS / 64, HDIM / 128), 256, 0, stream>>>(
      xnh, fc1_wt, fc1_b, MROWS, HDIM, DIM, nullptr, h1, nullptr);

  // 7) t2 = xn + h1 @ fc2_w + fc2_b   (reuse t1)
  k_gemm_wmma<EPI_RES><<<dim3(MROWS / 64, DIM / 128), 256, 0, stream>>>(
      h1, fc2_wt, fc2_b, MROWS, DIM, HDIM, t1, nullptr, xn);

  // 8) out = LN2(t2)
  k_layernorm<<<MROWS, 256, 0, stream>>>(t1, ln2_g, ln2_b, out, nullptr);
}
